// MultiHeadSelfAttention_6150393168648
// MI455X (gfx1250) — compile-verified, matrix operand order checked by reading, not run on hardware
//
#include <hip/hip_runtime.h>
#include <hip/hip_bf16.h>

// Relative-position MHA for MI455X (gfx1250), wave32 + WMMA bf16.
// B=2, T=512, D=512, H=8, HD=64.

typedef __bf16 bf16_t;
typedef __attribute__((ext_vector_type(16))) __bf16 v16bf;
typedef __attribute__((ext_vector_type(8)))  __bf16 v8bf;
typedef __attribute__((ext_vector_type(8)))  float  v8f;
typedef __attribute__((ext_vector_type(4)))  float  v4f;

#define DEV __device__ __forceinline__

static constexpr int B  = 2;
static constexpr int T  = 512;
static constexpr int D  = 512;
static constexpr int H  = 8;
static constexpr int HD = 64;

// ---------------------------------------------------------------------------
// WMMA operand loaders.
// A operand (16x32 bf16, M x K) per-lane layout: row m = lane%16, g = lane/16.
//   elements 0..7  = K in [8g, 8g+7]       (VGPRs 0..3, 2 packed per dword)
//   elements 8..15 = K in [16+8g, 16+8g+7] (VGPRs 4..7)
// B operand in [N,K]-major storage uses the IDENTICAL pattern (row = N).
// ---------------------------------------------------------------------------
DEV v16bf ld_tile_bf16(const bf16_t* base, int stride, int lane) {
  const int r = lane & 15, g = lane >> 4;
  const bf16_t* p = base + r * stride + g * 8;
  v8bf c0 = *(const v8bf*)(p);
  v8bf c1 = *(const v8bf*)(p + 16);
  v16bf o;
#pragma unroll
  for (int i = 0; i < 8; ++i) { o[i] = c0[i]; o[i + 8] = c1[i]; }
  return o;
}

// fp32 source, convert to bf16 in-flight (used for `inputs`).
DEV v16bf ld_tile_f32(const float* base, int stride, int lane) {
  const int r = lane & 15, g = lane >> 4;
  const float* p = base + r * stride + g * 8;
  v4f a0 = *(const v4f*)(p);
  v4f a1 = *(const v4f*)(p + 4);
  v4f b0 = *(const v4f*)(p + 16);
  v4f b1 = *(const v4f*)(p + 20);
  v16bf o;
#pragma unroll
  for (int i = 0; i < 4; ++i) {
    o[i]      = (bf16_t)a0[i];
    o[4 + i]  = (bf16_t)a1[i];
    o[8 + i]  = (bf16_t)b0[i];
    o[12 + i] = (bf16_t)b1[i];
  }
  return o;
}

// Non-temporal fp32 loader for the read-once 1 GB rel_pos_emb stream (TH=NT,
// keeps the 192 MB L2 free for the score RMW tiles and w slices).
DEV v16bf ld_tile_f32_nt(const float* base, int stride, int lane) {
  const int r = lane & 15, g = lane >> 4;
  const float* p = base + r * stride + g * 8;
  v4f a0 = __builtin_nontemporal_load((const v4f*)(p));
  v4f a1 = __builtin_nontemporal_load((const v4f*)(p + 4));
  v4f b0 = __builtin_nontemporal_load((const v4f*)(p + 16));
  v4f b1 = __builtin_nontemporal_load((const v4f*)(p + 20));
  v16bf o;
#pragma unroll
  for (int i = 0; i < 4; ++i) {
    o[i]      = (bf16_t)a0[i];
    o[4 + i]  = (bf16_t)a1[i];
    o[8 + i]  = (bf16_t)b0[i];
    o[12 + i] = (bf16_t)b1[i];
  }
  return o;
}

DEV v8f wmma_bf16(v16bf a, v16bf b, v8f c) {
  return __builtin_amdgcn_wmma_f32_16x16x32_bf16(false, a, false, b,
                                                 (short)0, c, false, false);
}

// ---------------------------------------------------------------------------
// Weight prep: transpose Wq/Wk/Wv/Wo to [N,K]-major bf16; Wp to plain bf16.
// ---------------------------------------------------------------------------
__global__ __launch_bounds__(256) void prep_weights(
    const float* __restrict__ Wq, const float* __restrict__ Wk,
    const float* __restrict__ Wv, const float* __restrict__ Wo,
    const float* __restrict__ Wp,
    bf16_t* __restrict__ Wqt, bf16_t* __restrict__ Wkt,
    bf16_t* __restrict__ Wvt, bf16_t* __restrict__ Wot,
    bf16_t* __restrict__ Wpb) {
  int idx = blockIdx.x * blockDim.x + threadIdx.x;
  if (idx >= D * D) return;
  int n = idx >> 9, k = idx & 511;
  int src = k * D + n;
  Wqt[idx] = (bf16_t)Wq[src];
  Wkt[idx] = (bf16_t)Wk[src];
  Wvt[idx] = (bf16_t)Wv[src];
  Wot[idx] = (bf16_t)Wo[src];
  Wpb[idx] = (bf16_t)Wp[idx];
}

// ---------------------------------------------------------------------------
// Q/K/V projection: [B*T, D] @ Wt^T + bias.
// mode 0: writes q_u = q + u_bias and q_v = q + v_bias, both [B,H,T,HD] bf16
// mode 1: writes k            [B,H,T,HD] bf16
// mode 2: writes v transposed [B,H,HD,T] bf16  (native B-operand for attn@V)
// ---------------------------------------------------------------------------
__global__ __launch_bounds__(128) void proj_kernel(
    const float* __restrict__ X, const bf16_t* __restrict__ Wt,
    const float* __restrict__ bias, const float* __restrict__ ubias,
    const float* __restrict__ vbias, bf16_t* __restrict__ out0,
    bf16_t* __restrict__ out1, int mode) {
  const int lane = threadIdx.x & 31, wave = threadIdx.x >> 5;
  const int m0 = blockIdx.x * 16;                 // over B*T
  const int n0 = (blockIdx.y * 4 + wave) * 16;    // over D
  v8f acc = {};
  for (int k0 = 0; k0 < D; k0 += 32)
    acc = wmma_bf16(ld_tile_f32(X + m0 * D + k0, D, lane),
                    ld_tile_bf16(Wt + n0 * D + k0, D, lane), acc);
  const int n = lane & 15, g = lane >> 4;
  const int ng = n0 + n;
  const int h = ng >> 6, hd = ng & 63;
  const float bv = bias[ng];
  if (mode == 0) {
    const float ub = ubias[ng], vb = vbias[ng];
#pragma unroll
    for (int v = 0; v < 8; ++v) {
      int m = m0 + v + 8 * g;
      int bb = m >> 9, t = m & 511;
      float val = acc[v] + bv;
      int idx = ((bb * H + h) * T + t) * HD + hd;
      out0[idx] = (bf16_t)(val + ub);
      out1[idx] = (bf16_t)(val + vb);
    }
  } else if (mode == 1) {
#pragma unroll
    for (int v = 0; v < 8; ++v) {
      int m = m0 + v + 8 * g;
      int bb = m >> 9, t = m & 511;
      out0[((bb * H + h) * T + t) * HD + hd] = (bf16_t)(acc[v] + bv);
    }
  } else {
#pragma unroll
    for (int v = 0; v < 8; ++v) {
      int m = m0 + v + 8 * g;
      int bb = m >> 9, t = m & 511;
      out0[((bb * H + h) * HD + hd) * T + t] = (bf16_t)(acc[v] + bv);
    }
  }
}

// ---------------------------------------------------------------------------
// w[b,t,h,k] = sum_hd q_v[b,h,t,hd] * Wp[k, h*64+hd]   (per-head Wp^T @ q_v)
// ---------------------------------------------------------------------------
__global__ __launch_bounds__(128) void wproj_kernel(
    const bf16_t* __restrict__ qv, const bf16_t* __restrict__ Wpb,
    bf16_t* __restrict__ w) {
  const int lane = threadIdx.x & 31, wave = threadIdx.x >> 5;
  const int m0 = blockIdx.x * 16;                 // over B*T
  const int n0 = (blockIdx.y * 4 + wave) * 16;    // over D (k of w)
  const int h  = blockIdx.z;
  const int bb = m0 >> 9, t0 = m0 & 511;
  const bf16_t* abase = qv + ((bb * H + h) * T + t0) * HD;
  const bf16_t* bbase = Wpb + n0 * D + h * HD;    // Bt[n,kk] = Wp[n, h*64+kk]
  v8f acc = {};
#pragma unroll
  for (int k0 = 0; k0 < HD; k0 += 32)
    acc = wmma_bf16(ld_tile_bf16(abase + k0, HD, lane),
                    ld_tile_bf16(bbase + k0, D, lane), acc);
  const int n = lane & 15, g = lane >> 4;
#pragma unroll
  for (int v = 0; v < 8; ++v) {
    int t = t0 + v + 8 * g;
    w[((bb * T + t) * H + h) * D + n0 + n] = (bf16_t)acc[v];
  }
}

// ---------------------------------------------------------------------------
// qk scores: scores[b,h,i,j] = sum_d q_u[b,h,i,d] * k[b,h,j,d]  (fp32 out)
// ---------------------------------------------------------------------------
__global__ __launch_bounds__(128) void qk_kernel(
    const bf16_t* __restrict__ qu, const bf16_t* __restrict__ kk,
    float* __restrict__ scores) {
  const int lane = threadIdx.x & 31, wave = threadIdx.x >> 5;
  const int bh = blockIdx.x;
  const int i0 = blockIdx.y * 16;
  const int j0 = (blockIdx.z * 4 + wave) * 16;
  const bf16_t* abase = qu + (bh * T + i0) * HD;
  const bf16_t* bbase = kk + (bh * T + j0) * HD;
  v8f acc = {};
#pragma unroll
  for (int k0 = 0; k0 < HD; k0 += 32)
    acc = wmma_bf16(ld_tile_bf16(abase + k0, HD, lane),
                    ld_tile_bf16(bbase + k0, HD, lane), acc);
  const int n = lane & 15, g = lane >> 4;
  float* o = scores + (bh * T + i0) * T + j0 + n;
#pragma unroll
  for (int v = 0; v < 8; ++v) o[(v + 8 * g) * T] = acc[v];
}

// ---------------------------------------------------------------------------
// pos scores accumulated into qk scores. The dominant (1.07 GB read) kernel.
// Per (b, i, j-tile): D_tile[m=j_local, n=h] = R[b,i,j,:] x w[b,i,h,:]^T.
// M dim = j (dense, non-temporal fp32->bf16 stream); N dim = h padded 8->16.
// The 8x512 w[b,i] slice is staged zero-padded to 16x512 in LDS once per
// block (8 waves / 8 j-tiles share it), so the B operand is an unpredicated
// ds_load and the inner loop carries only the R stream on the vmem path.
// C operand = previously written qk scores (read-modify-write via WMMA C).
// ---------------------------------------------------------------------------
__global__ __launch_bounds__(256) void pos_kernel(
    const float* __restrict__ R, const bf16_t* __restrict__ w,
    float* __restrict__ scores) {
  __shared__ bf16_t wl[16 * D];                   // 16 KB, rows 8..15 = zeros
  const int tid = threadIdx.x;
  const int lane = tid & 31, wave = tid >> 5;
  const int bi = blockIdx.x;                      // 0 .. B*T-1
  const int bb = bi >> 9, i = bi & 511;
  const int j0 = (blockIdx.y * 8 + wave) * 16;
  const int n = lane & 15, g = lane >> 4;

  // Cooperative fill: 8 valid h rows (4096 elems) + 8 zero rows.
  {
    const bf16_t* wsrc = w + (bb * T + i) * H * D;
    const int e = tid * 16;                       // 16 elems per thread
    *(v8bf*)(wl + e)     = *(const v8bf*)(wsrc + e);
    *(v8bf*)(wl + e + 8) = *(const v8bf*)(wsrc + e + 8);
    v8bf z = {};
    *(v8bf*)(wl + H * D + e)     = z;
    *(v8bf*)(wl + H * D + e + 8) = z;
  }
  __syncthreads();

  // C tile: element (m=j_local, n=h) -> scores[b, h, i, j0+m]; m = v + 8g.
  v8f acc;
  {
    const int hc = n < 8 ? n : 7;
    const float* cbase = scores + ((bb * H + hc) * T + i) * T + j0 + 8 * g;
    v4f c0 = *(const v4f*)(cbase);
    v4f c1 = *(const v4f*)(cbase + 4);
    const bool ok = n < 8;
#pragma unroll
    for (int v = 0; v < 4; ++v) {
      acc[v]     = ok ? c0[v] : 0.0f;
      acc[4 + v] = ok ? c1[v] : 0.0f;
    }
  }

  const float* abase = R + ((bb * T + i) * T + j0) * D;   // rows = j, native
  for (int k0 = 0; k0 < D; k0 += 32)
    acc = wmma_bf16(ld_tile_f32_nt(abase + k0, D, lane),
                    ld_tile_bf16(wl + k0, D, lane), acc);

  if (n < 8) {
    float* obase = scores + ((bb * H + n) * T + i) * T + j0 + 8 * g;
    v4f o0 = {acc[0], acc[1], acc[2], acc[3]};
    v4f o1 = {acc[4], acc[5], acc[6], acc[7]};
    *(v4f*)(obase)     = o0;
    *(v4f*)(obase + 4) = o1;
  }
}

// ---------------------------------------------------------------------------
// Softmax over j with 1/sqrt(HD)=0.125 scaling; one wave per row; bf16 out.
// (The bp . q_v term is constant over j and cancels here -> dropped exactly.)
// ---------------------------------------------------------------------------
__global__ __launch_bounds__(128) void softmax_kernel(
    const float* __restrict__ scores, bf16_t* __restrict__ attn) {
  const int lane = threadIdx.x & 31, wave = threadIdx.x >> 5;
  const int row = blockIdx.x * 4 + wave;          // 0 .. B*H*T-1
  const float* p = scores + row * T + lane * 16;
  float x[16];
#pragma unroll
  for (int c = 0; c < 4; ++c) {
    v4f v = *(const v4f*)(p + c * 4);
#pragma unroll
    for (int e = 0; e < 4; ++e) x[c * 4 + e] = v[e] * 0.125f;
  }
  float mx = x[0];
#pragma unroll
  for (int e = 1; e < 16; ++e) mx = fmaxf(mx, x[e]);
#pragma unroll
  for (int s = 16; s >= 1; s >>= 1) mx = fmaxf(mx, __shfl_xor(mx, s, 32));
  float sum = 0.0f;
#pragma unroll
  for (int e = 0; e < 16; ++e) { x[e] = __expf(x[e] - mx); sum += x[e]; }
#pragma unroll
  for (int s = 16; s >= 1; s >>= 1) sum += __shfl_xor(sum, s, 32);
  const float inv = 1.0f / sum;
  bf16_t* o = attn + row * T + lane * 16;
  v8bf o0, o1;
#pragma unroll
  for (int e = 0; e < 8; ++e) {
    o0[e] = (bf16_t)(x[e] * inv);
    o1[e] = (bf16_t)(x[8 + e] * inv);
  }
  *(v8bf*)(o)     = o0;
  *(v8bf*)(o + 8) = o1;
}

// ---------------------------------------------------------------------------
// context[b,t,h*64+d] = sum_j attn[b,h,t,j] * v[b,h,j,d]   (vt is [B,H,HD,T])
// ---------------------------------------------------------------------------
__global__ __launch_bounds__(128) void ctx_kernel(
    const bf16_t* __restrict__ attn, const bf16_t* __restrict__ vt,
    bf16_t* __restrict__ ctx) {
  const int lane = threadIdx.x & 31, wave = threadIdx.x >> 5;  // wave = d-tile
  const int bh = blockIdx.x;
  const int i0 = blockIdx.y * 16;
  const int d0 = wave * 16;
  const int bb = bh >> 3, h = bh & 7;
  const bf16_t* abase = attn + (bh * T + i0) * T;
  const bf16_t* bbase = vt + (bh * HD + d0) * T;
  v8f acc = {};
  for (int k0 = 0; k0 < T; k0 += 32)
    acc = wmma_bf16(ld_tile_bf16(abase + k0, T, lane),
                    ld_tile_bf16(bbase + k0, T, lane), acc);
  const int n = lane & 15, g = lane >> 4;
#pragma unroll
  for (int v = 0; v < 8; ++v) {
    int t = i0 + v + 8 * g;
    ctx[(bb * T + t) * D + h * HD + d0 + n] = (bf16_t)acc[v];
  }
}

// ---------------------------------------------------------------------------
// out = context @ Wo + bo  (fp32 output)
// ---------------------------------------------------------------------------
__global__ __launch_bounds__(128) void out_kernel(
    const bf16_t* __restrict__ ctx, const bf16_t* __restrict__ Wot,
    const float* __restrict__ bo, float* __restrict__ out) {
  const int lane = threadIdx.x & 31, wave = threadIdx.x >> 5;
  const int m0 = blockIdx.x * 16;
  const int n0 = (blockIdx.y * 4 + wave) * 16;
  v8f acc = {};
  for (int k0 = 0; k0 < D; k0 += 32)
    acc = wmma_bf16(ld_tile_bf16(ctx + m0 * D + k0, D, lane),
                    ld_tile_bf16(Wot + n0 * D + k0, D, lane), acc);
  const int n = lane & 15, g = lane >> 4;
  const float bias = bo[n0 + n];
  float* o = out + m0 * D + n0 + n;
#pragma unroll
  for (int v = 0; v < 8; ++v) o[(v + 8 * g) * D] = acc[v] + bias;
}

// ---------------------------------------------------------------------------
extern "C" void kernel_launch(void* const* d_in, const int* in_sizes, int n_in,
                              void* d_out, int out_size, void* d_ws,
                              size_t ws_size, hipStream_t stream) {
  const float* X  = (const float*)d_in[0];   // inputs [B,T,D]
  const float* R  = (const float*)d_in[1];   // rel_pos_emb [B,T,T,D]
  const float* Wq = (const float*)d_in[2];
  const float* bq = (const float*)d_in[3];
  const float* Wk = (const float*)d_in[4];
  const float* bk = (const float*)d_in[5];
  const float* Wv = (const float*)d_in[6];
  const float* bv = (const float*)d_in[7];
  const float* Wp = (const float*)d_in[8];
  // d_in[9] = bp: constant over j -> cancels in softmax, unused.
  const float* Wo = (const float*)d_in[10];
  const float* bo = (const float*)d_in[11];
  const float* ub = (const float*)d_in[12];  // [H*HD]
  const float* vb = (const float*)d_in[13];
  float* out = (float*)d_out;

  char* ws = (char*)d_ws;
  auto alloc = [&](size_t bytes) -> char* {
    char* p = ws;
    ws += (bytes + 255) & ~(size_t)255;
    return p;
  };
  const size_t NE  = (size_t)B * H * T * HD;     // 524288
  bf16_t* q_u  = (bf16_t*)alloc(NE * 2);
  bf16_t* q_v  = (bf16_t*)alloc(NE * 2);
  bf16_t* k_bf = (bf16_t*)alloc(NE * 2);
  bf16_t* v_t  = (bf16_t*)alloc(NE * 2);
  bf16_t* w_bf = (bf16_t*)alloc((size_t)B * T * H * D * 2);
  bf16_t* attn = (bf16_t*)alloc((size_t)B * H * T * T * 2);
  bf16_t* ctx  = (bf16_t*)alloc((size_t)B * T * D * 2);
  float*  scr  = (float*)alloc((size_t)B * H * T * T * 4);
  bf16_t* Wqt  = (bf16_t*)alloc((size_t)D * D * 2);
  bf16_t* Wkt  = (bf16_t*)alloc((size_t)D * D * 2);
  bf16_t* Wvt  = (bf16_t*)alloc((size_t)D * D * 2);
  bf16_t* Wot  = (bf16_t*)alloc((size_t)D * D * 2);
  bf16_t* Wpb  = (bf16_t*)alloc((size_t)D * D * 2);

  prep_weights<<<dim3((D * D + 255) / 256), 256, 0, stream>>>(
      Wq, Wk, Wv, Wo, Wp, Wqt, Wkt, Wvt, Wot, Wpb);

  // Projections: grid (B*T/16, D/16/4), 4 waves per block.
  proj_kernel<<<dim3(64, 8), 128, 0, stream>>>(X, Wqt, bq, ub, vb, q_u, q_v, 0);
  proj_kernel<<<dim3(64, 8), 128, 0, stream>>>(X, Wkt, bk, ub, vb, k_bf, k_bf, 1);
  proj_kernel<<<dim3(64, 8), 128, 0, stream>>>(X, Wvt, bv, ub, vb, v_t, v_t, 2);

  wproj_kernel<<<dim3(64, 8, H), 128, 0, stream>>>(q_v, Wpb, w_bf);

  qk_kernel<<<dim3(B * H, T / 16, 8), 128, 0, stream>>>(q_u, k_bf, scr);

  // 8 waves/block share one LDS-staged w[b,i]; grid (B*T, T/16/8).
  pos_kernel<<<dim3(B * T, 4), 256, 0, stream>>>(R, w_bf, scr);

  softmax_kernel<<<dim3(B * H * T / 4), 128, 0, stream>>>(scr, attn);

  ctx_kernel<<<dim3(B * H, T / 16), 128, 0, stream>>>(attn, v_t, ctx);

  out_kernel<<<dim3(64, 8), 128, 0, stream>>>(ctx, Wot, bo, out);
}
